// SA_upsample_43310450213254
// MI455X (gfx1250) — matrix operand order repaired and checked by reading, not compiled, not run on hardware
//
#include <hip/hip_runtime.h>
#include <hip/hip_bf16.h>

// ---------------- problem constants ----------------
constexpr int B_  = 4;
constexpr int C_  = 256;
constexpr int H_  = 48,  W_  = 48;
constexpr int N_  = H_ * W_;          // 2304
constexpr int HL_ = 96,  WL_ = 96;
constexpr int NL_ = HL_ * WL_;        // 9216
constexpr int KD_ = 128;
constexpr int VD_ = 256;
constexpr int OD_ = 256;

// ---------------- workspace layout (bytes) ----------------
constexpr size_t OFF_XL2 = 0;                                   // bf16 [B][NL][C]
constexpr size_t SZ_XL2  = (size_t)B_ * NL_ * C_ * 2;
constexpr size_t OFF_XBF = OFF_XL2 + SZ_XL2;                    // bf16 [B][N][C]
constexpr size_t SZ_XBF  = (size_t)B_ * N_ * C_ * 2;
constexpr size_t OFF_WK  = OFF_XBF + SZ_XBF;                    // bf16 [KD][C]
constexpr size_t SZ_WK   = (size_t)KD_ * C_ * 2;
constexpr size_t OFF_WV  = OFF_WK + SZ_WK;                      // bf16 [VD][C]
constexpr size_t SZ_WV   = (size_t)VD_ * C_ * 2;
constexpr size_t OFF_WR  = OFF_WV + SZ_WV;                      // bf16 [OD][VD]
constexpr size_t SZ_WR   = (size_t)OD_ * VD_ * 2;
constexpr size_t OFF_Q   = OFF_WR + SZ_WR;                      // bf16 [B][NL][KD]
constexpr size_t SZ_Q    = (size_t)B_ * NL_ * KD_ * 2;
constexpr size_t OFF_KT  = OFF_Q + SZ_Q;                        // bf16 [B][N][KD]
constexpr size_t SZ_KT   = (size_t)B_ * N_ * KD_ * 2;
constexpr size_t OFF_V   = OFF_KT + SZ_KT;                      // bf16 [B][VD][N]
constexpr size_t SZ_V    = (size_t)B_ * VD_ * N_ * 2;
constexpr size_t OFF_Z   = OFF_V + SZ_V;                        // bf16 [B][NL][VD]
constexpr size_t SZ_Z    = (size_t)B_ * NL_ * VD_ * 2;

// ---------------- types ----------------
typedef __bf16 v16bf __attribute__((ext_vector_type(16)));
typedef float  v8f   __attribute__((ext_vector_type(8)));

union AF {            // one 16x32 bf16 WMMA operand (A or B fragment), 8 VGPRs
    v16bf v;
    unsigned int   d[8];
    unsigned short u[16];
};

#define DEVINL __device__ __forceinline__

DEVINL unsigned short f2bf(float f) {
    unsigned u = __float_as_uint(f);
    u = u + 0x7FFFu + ((u >> 16) & 1u);   // round-to-nearest-even
    return (unsigned short)(u >> 16);
}
DEVINL float bf2f(unsigned short h) { return __uint_as_float(((unsigned)h) << 16); }

DEVINL v8f vzero() {
    v8f z;
#pragma unroll
    for (int i = 0; i < 8; ++i) z[i] = 0.0f;
    return z;
}

DEVINL v8f wmma_bf16(const AF& a, const AF& b, v8f c) {
    return __builtin_amdgcn_wmma_f32_16x16x32_bf16(
        /*neg_a=*/false, a.v, /*neg_b=*/false, b.v,
        /*c_mod=*/(short)0, c, /*reuse_a=*/false, /*reuse_b=*/false);
}

// A-fragment K-pair base for VGPR g, lane-half hf (16-bit A 16x32 layout, ISA 7.12.2)
DEVINL int a_kbase(int g, int hf) { return (g < 4 ? 2 * g : 16 + 2 * (g - 4)) + 8 * hf; }
// B-fragment K-pair base for VGPR g, lane-half hf (same K order as A fragment)
DEVINL int b_kbase(int g, int hf) { return (g < 4 ? 2 * g : 16 + 2 * (g - 4)) + 8 * hf; }

// ---------------- kernel 1: bilinear upsample (align_corners) + alpha*xl -> bf16 [B][NL][C] ----------------
__global__ void __launch_bounds__(256) k_upsample(const float* __restrict__ x,   // [B][C][H][W]
                                                  const float* __restrict__ xl,  // [B][C][HL][WL]
                                                  const float* __restrict__ alpha,
                                                  unsigned short* __restrict__ xl2) {
    int idx = blockIdx.x * blockDim.x + threadIdx.x;   // c fastest -> coalesced writes
    if (idx >= B_ * NL_ * C_) return;
    int c = idx % C_;
    int q = (idx / C_) % NL_;
    int b = idx / (C_ * NL_);
    int qx = q % WL_, qy = q / WL_;
    float fy = qy * (float)(H_ - 1) / (float)(HL_ - 1);
    float fx = qx * (float)(W_ - 1) / (float)(WL_ - 1);
    int y0 = (int)floorf(fy), x0 = (int)floorf(fx);
    int y1 = (y0 + 1 < H_) ? y0 + 1 : H_ - 1;
    int x1 = (x0 + 1 < W_) ? x0 + 1 : W_ - 1;
    float wy = fy - (float)y0, wx = fx - (float)x0;
    const float* xb = x + ((size_t)b * C_ + c) * (H_ * W_);
    float v00 = xb[y0 * W_ + x0], v01 = xb[y0 * W_ + x1];
    float v10 = xb[y1 * W_ + x0], v11 = xb[y1 * W_ + x1];
    float top = v00 * (1.f - wx) + v01 * wx;
    float bot = v10 * (1.f - wx) + v11 * wx;
    float val = top * (1.f - wy) + bot * wy + alpha[0] * xl[((size_t)b * C_ + c) * NL_ + q];
    xl2[idx] = f2bf(val);
}

// ---------------- kernel 2a: f32 -> bf16 flat (weights) ----------------
__global__ void __launch_bounds__(256) k_cvt(const float* __restrict__ src,
                                             unsigned short* __restrict__ dst, int n) {
    int i = blockIdx.x * blockDim.x + threadIdx.x;
    if (i < n) dst[i] = f2bf(src[i]);
}

// ---------------- kernel 2b: x [B][C][N] f32 -> bf16 [B][N][C] (transpose) ----------------
__global__ void __launch_bounds__(256) k_cvt_xpose(const float* __restrict__ src,
                                                   unsigned short* __restrict__ dst) {
    int idx = blockIdx.x * blockDim.x + threadIdx.x;   // c fastest -> coalesced writes
    if (idx >= B_ * N_ * C_) return;
    int c = idx % C_;
    int n = (idx / C_) % N_;
    int b = idx / (C_ * N_);
    dst[idx] = f2bf(src[((size_t)b * C_ + c) * N_ + n]);
}

// ---------------- kernel 3: WMMA GEMM  Out[o,n] = act( sum_c W[o,c] * X[n,c] ) ----------------
// X is row-major in the contraction dim -> B fragments are contiguous b128 loads.
// MODE: 0 = +bias (value conv), 1 = BN + leaky_relu
// TRANS: 0 -> store [B][O][Nn], 1 -> store [B][Nn][O] (row-major in O, for Q / K^T)
// F32OUT: 0 -> bf16 out, 1 -> f32 out
template <int MODE, int TRANS, int F32OUT>
__global__ void __launch_bounds__(32)
k_gemm(const unsigned short* __restrict__ Wb,   // bf16 [O][Cc]
       const unsigned short* __restrict__ Xb,   // bf16 [B][Nn][Cc]
       void* __restrict__ outp,
       const float* __restrict__ gam, const float* __restrict__ bet,
       const float* __restrict__ mu,  const float* __restrict__ var,
       const float* __restrict__ bias,
       int O, int Cc, int Nn, int ntiles /* Nn/64 */) {
    int lane = threadIdx.x;
    int l15 = lane & 15, hf = lane >> 4;
    int blk  = blockIdx.x;
    int nblk = blk % ntiles;
    int t    = blk / ntiles;
    int ot   = t % (O / 16);
    int b    = t / (O / 16);
    int n0    = nblk * 64;
    int orow0 = ot * 16;
    const unsigned short* X = Xb + (size_t)b * Nn * Cc;

    v8f acc[4];
#pragma unroll
    for (int s = 0; s < 4; ++s) acc[s] = vzero();

    for (int ks = 0; ks < Cc; ks += 32) {
        AF a;
        int orow = orow0 + l15;
#pragma unroll
        for (int g = 0; g < 8; ++g)   // W row-major in c: two contiguous b128 groups
            a.d[g] = *(const unsigned int*)(Wb + (size_t)orow * Cc + ks + a_kbase(g, hf));
#pragma unroll
        for (int s = 0; s < 4; ++s) {
            AF bf;
            int ncol = n0 + s * 16 + l15;
#pragma unroll
            for (int g = 0; g < 8; ++g)   // X row-major in c: two contiguous b128 groups
                bf.d[g] = *(const unsigned int*)(X + (size_t)ncol * Cc + ks + b_kbase(g, hf));
            acc[s] = wmma_bf16(a, bf, acc[s]);
        }
    }

#pragma unroll
    for (int s = 0; s < 4; ++s) {
        int ncol = n0 + s * 16 + l15;
#pragma unroll
        for (int g = 0; g < 8; ++g) {
            int orow = orow0 + g + 8 * hf;   // D-tile row (output channel)
            float v = acc[s][g];
            if (MODE == 0) {
                v += bias[orow];
            } else {
                float sc = gam[orow] * rsqrtf(var[orow] + 1e-5f);
                v = (v - mu[orow]) * sc + bet[orow];
                v = (v >= 0.f) ? v : 0.01f * v;
            }
            size_t off = TRANS ? ((size_t)b * Nn * O + (size_t)ncol * O + orow)
                               : ((size_t)b * O * Nn + (size_t)orow * Nn + ncol);
            if (F32OUT) ((float*)outp)[off] = v;
            else        ((unsigned short*)outp)[off] = f2bf(v);
        }
    }
}

// ---------------- kernel 4: fused flash attention + (xl2 + beta*context) -> Z bf16 ----------------
// One wave handles 16 query rows, full Kd=128 and Vd=256, streaming keys 32 at a time.
__global__ void __launch_bounds__(32)
k_attn(const unsigned short* __restrict__ Q,    // bf16 [B][NL][KD]
       const unsigned short* __restrict__ Kt,   // bf16 [B][N][KD]
       const unsigned short* __restrict__ V,    // bf16 [B][VD][N]
       const unsigned short* __restrict__ xl2,  // bf16 [B][NL][C], C==VD
       const float* __restrict__ betap,
       unsigned short* __restrict__ Z) {        // bf16 [B][NL][VD]
    int lane = threadIdx.x;
    int l15 = lane & 15, hf = lane >> 4;
    int qt = blockIdx.x % (NL_ / 16);
    int b  = blockIdx.x / (NL_ / 16);
    const unsigned short* Qb = Q  + ((size_t)b * NL_ + (size_t)qt * 16) * KD_;
    const unsigned short* Kb = Kt + (size_t)b * N_ * KD_;
    const unsigned short* Vb = V  + (size_t)b * VD_ * N_;

    // Q fragments (A): 4 over Kd=128, row = query l15, pairs contiguous in KD
    AF qf[4];
#pragma unroll
    for (int ks = 0; ks < 4; ++ks)
#pragma unroll
        for (int g = 0; g < 8; ++g)
            qf[ks].d[g] = *(const unsigned int*)(Qb + (size_t)l15 * KD_ + ks * 32 + a_kbase(g, hf));

    v8f acc[16];                 // 16 Vd-tiles of 16x16 f32
#pragma unroll
    for (int t = 0; t < 16; ++t) acc[t] = vzero();
    float mrow[8], lrow[8];
#pragma unroll
    for (int g = 0; g < 8; ++g) { mrow[g] = -1e30f; lrow[g] = 0.f; }
    const float rs = 0.08838834764831845f;   // Kd^-0.5, Kd=128

    __shared__ unsigned short pb[16 * 32];   // P staging: [16 q rows][32 key cols] bf16

    for (int nb = 0; nb < N_; nb += 32) {
        if (nb + 32 < N_)   // prefetch next key block (global_prefetch_b8)
            __builtin_prefetch(Kb + (size_t)(nb + 32 + l15) * KD_, 0, 0);

        // ---- S = Q @ K^T for 32 keys (two 16-col groups) ----
        v8f s0 = vzero(), s1 = vzero();
#pragma unroll
        for (int ks = 0; ks < 4; ++ks) {
            AF k0, k1;
#pragma unroll
            for (int g = 0; g < 8; ++g) {
                int kd = ks * 32 + b_kbase(g, hf);
                k0.d[g] = *(const unsigned int*)(Kb + (size_t)(nb + l15) * KD_ + kd);
                k1.d[g] = *(const unsigned int*)(Kb + (size_t)(nb + 16 + l15) * KD_ + kd);
            }
            s0 = wmma_bf16(qf[ks], k0, s0);
            s1 = wmma_bf16(qf[ks], k1, s1);
        }

        // ---- online softmax (row = g + 8*hf, cols spread over 16 lanes) ----
        float p0[8], p1[8], sc[8];
#pragma unroll
        for (int g = 0; g < 8; ++g) {
            float a = s0[g] * rs, c = s1[g] * rs;
            p0[g] = a; p1[g] = c;
            float t = fmaxf(a, c);
#pragma unroll
            for (int m = 1; m < 16; m <<= 1) t = fmaxf(t, __shfl_xor(t, m, 16));
            float nm = fmaxf(mrow[g], t);
            sc[g] = __expf(mrow[g] - nm);
            mrow[g] = nm;
            p0[g] = __expf(p0[g] - nm);
            p1[g] = __expf(p1[g] - nm);
            float rsum = p0[g] + p1[g];
#pragma unroll
            for (int m = 1; m < 16; m <<= 1) rsum += __shfl_xor(rsum, m, 16);
            lrow[g] = lrow[g] * sc[g] + rsum;
        }
#pragma unroll
        for (int t = 0; t < 16; ++t)
#pragma unroll
            for (int g = 0; g < 8; ++g) acc[t][g] *= sc[g];

        // ---- P (C-layout) -> LDS -> A-fragment ----
#pragma unroll
        for (int g = 0; g < 8; ++g) {
            pb[(g + 8 * hf) * 32 + l15]      = f2bf(p0[g]);
            pb[(g + 8 * hf) * 32 + 16 + l15] = f2bf(p1[g]);
        }
        __syncthreads();
        AF pf;
#pragma unroll
        for (int g = 0; g < 8; ++g)
            pf.d[g] = *(const unsigned int*)&pb[l15 * 32 + a_kbase(g, hf)];
        __syncthreads();

        // ---- O += P @ V over all 16 Vd-tiles ----
#pragma unroll
        for (int t = 0; t < 16; ++t) {
            AF vf;
#pragma unroll
            for (int g = 0; g < 8; ++g) {
                int nn = nb + b_kbase(g, hf);
                vf.d[g] = *(const unsigned int*)(Vb + (size_t)(t * 16 + l15) * N_ + nn);
            }
            acc[t] = wmma_bf16(pf, vf, acc[t]);
        }
    }

    // ---- epilogue: Z[q][v] = xl2[q][v] + beta * (O / l), both [NL][256] ----
    float beta = betap[0];
#pragma unroll
    for (int t = 0; t < 16; ++t) {
#pragma unroll
        for (int g = 0; g < 8; ++g) {
            int vch = t * 16 + l15;              // D-tile col = value channel
            int qpos = qt * 16 + g + 8 * hf;     // D-tile row = query position
            float ctx = acc[t][g] / lrow[g];
            size_t off = ((size_t)b * NL_ + qpos) * VD_ + vch;   // same index for xl2 (C==VD)
            Z[off] = f2bf(bf2f(xl2[off]) + beta * ctx);
        }
    }
}

// ---------------- host launch ----------------
extern "C" void kernel_launch(void* const* d_in, const int* in_sizes, int n_in,
                              void* d_out, int out_size, void* d_ws, size_t ws_size,
                              hipStream_t stream) {
    const float* x      = (const float*)d_in[0];
    const float* xl     = (const float*)d_in[1];
    const float* Wk     = (const float*)d_in[2];
    const float* bnk_g  = (const float*)d_in[3];
    const float* bnk_b  = (const float*)d_in[4];
    const float* bnk_m  = (const float*)d_in[5];
    const float* bnk_v  = (const float*)d_in[6];
    const float* Wv     = (const float*)d_in[7];
    const float* bv     = (const float*)d_in[8];
    const float* Wr     = (const float*)d_in[9];
    const float* bnr_g  = (const float*)d_in[10];
    const float* bnr_b  = (const float*)d_in[11];
    const float* bnr_m  = (const float*)d_in[12];
    const float* bnr_v  = (const float*)d_in[13];
    const float* alpha  = (const float*)d_in[14];
    const float* beta   = (const float*)d_in[15];

    char* ws = (char*)d_ws;
    unsigned short* xl2b = (unsigned short*)(ws + OFF_XL2);
    unsigned short* xbf  = (unsigned short*)(ws + OFF_XBF);
    unsigned short* wkb  = (unsigned short*)(ws + OFF_WK);
    unsigned short* wvb  = (unsigned short*)(ws + OFF_WV);
    unsigned short* wrb  = (unsigned short*)(ws + OFF_WR);
    unsigned short* Qb   = (unsigned short*)(ws + OFF_Q);
    unsigned short* Ktb  = (unsigned short*)(ws + OFF_KT);
    unsigned short* Vb   = (unsigned short*)(ws + OFF_V);
    unsigned short* Zb   = (unsigned short*)(ws + OFF_Z);

    // 1) xl2 = bilinear(x) + alpha*xl  (bf16, [B][NL][C])
    {
        int n = B_ * NL_ * C_;
        k_upsample<<<(n + 255) / 256, 256, 0, stream>>>(x, xl, alpha, xl2b);
    }
    // 2) bf16 conversions (x transposed to [B][N][C]; weights flat)
    {
        int n = B_ * N_ * C_;
        k_cvt_xpose<<<(n + 255) / 256, 256, 0, stream>>>(x, xbf);
        n = KD_ * C_;  k_cvt<<<(n + 255) / 256, 256, 0, stream>>>(Wk, wkb, n);
        n = VD_ * C_;  k_cvt<<<(n + 255) / 256, 256, 0, stream>>>(Wv, wvb, n);
        n = OD_ * VD_; k_cvt<<<(n + 255) / 256, 256, 0, stream>>>(Wr, wrb, n);
    }
    // 3) K^T = bn_lrelu(Wk @ x) stored [B][N][KD]
    k_gemm<1, 1, 0><<<B_ * (KD_ / 16) * (N_ / 64), 32, 0, stream>>>(
        wkb, xbf, Ktb, bnk_g, bnk_b, bnk_m, bnk_v, nullptr, KD_, C_, N_, N_ / 64);
    // 4) Q = bn_lrelu(Wk @ xl2) stored [B][NL][KD]
    k_gemm<1, 1, 0><<<B_ * (KD_ / 16) * (NL_ / 64), 32, 0, stream>>>(
        wkb, xl2b, Qb, bnk_g, bnk_b, bnk_m, bnk_v, nullptr, KD_, C_, NL_, NL_ / 64);
    // 5) V = Wv @ x + bv stored [B][VD][N]
    k_gemm<0, 0, 0><<<B_ * (VD_ / 16) * (N_ / 64), 32, 0, stream>>>(
        wvb, xbf, Vb, nullptr, nullptr, nullptr, nullptr, bv, VD_, C_, N_, N_ / 64);
    // 6) fused flash attention -> Z = xl2 + beta*context (bf16, [B][NL][VD])
    k_attn<<<B_ * (NL_ / 16), 32, 0, stream>>>(Qb, Ktb, Vb, xl2b, beta, Zb);
    // 7) out = bn_lrelu(Wr @ Z) -> f32 [B][OD][HL][WL]
    k_gemm<1, 0, 1><<<B_ * (OD_ / 16) * (NL_ / 64), 32, 0, stream>>>(
        wrb, Zb, d_out, bnr_g, bnr_b, bnr_m, bnr_v, nullptr, OD_, VD_, NL_, NL_ / 64);
}
